// DownsampleCIF_7155415515215
// MI455X (gfx1250) — matrix operand order repaired and checked by reading, not compiled, not run on hardware
//
#include <hip/hip_runtime.h>

// ---------------- problem constants ----------------
#define BB 4
#define SS 2048
#define DD 256
#define HH 256
#define TT 512          // T_OUT = S / DS
#define K1 8192         // 32 * D
#define K2 768          // 3 * H

typedef __bf16 bf16_t;
typedef __attribute__((ext_vector_type(16))) __bf16 v16bf;
typedef __attribute__((ext_vector_type(8)))  __bf16 v8bf;
typedef __attribute__((ext_vector_type(8)))  float  v8f;

__device__ inline v16bf ld_frag2(const bf16_t* p0, const bf16_t* p1) {
    v8bf lo = *(const v8bf*)p0;
    v8bf hi = *(const v8bf*)p1;
    return __builtin_shufflevector(lo, hi, 0,1,2,3,4,5,6,7,8,9,10,11,12,13,14,15);
}

__device__ inline v8f wmma_bf16(v16bf a, v16bf b, v8f c) {
    return __builtin_amdgcn_wmma_f32_16x16x32_bf16(false, a, false, b, (short)0, c, false, false);
}

// ---------------- kernel 0: pack B matrices to bf16 [kb][n][32] ----------------
__global__ __launch_bounds__(256) void k_pack(const float* __restrict__ projw,
                                              const float* __restrict__ c1w,
                                              bf16_t* __restrict__ B1,
                                              bf16_t* __restrict__ B2) {
    int idx = blockIdx.x * 256 + threadIdx.x;
    if (idx < K1 * HH) {
        int k = idx >> 8, n = idx & 255;                     // proj_w[k][n], row-major (8192,256)
        B1[(((size_t)(k >> 5)) * HH + n) * 32 + (k & 31)] = (bf16_t)projw[(size_t)k * HH + n];
    } else {
        int i2 = idx - K1 * HH;
        if (i2 < K2 * HH) {
            int k = i2 >> 8, n = i2 & 255;                   // k = j*256 + hin, n = h_out
            int j = k >> 8, hin = k & 255;                   // conv1d_w shape (H, H, 3)
            B2[(((size_t)(k >> 5)) * HH + n) * 32 + (k & 31)] =
                (bf16_t)c1w[((size_t)n * HH + hin) * 3 + j];
        }
    }
}

// ---------------- zero the CIF output region ----------------
__global__ __launch_bounds__(256) void k_zero(float* __restrict__ p, int n) {
    int i = blockIdx.x * 256 + threadIdx.x;
    if (i < n) p[i] = 0.f;
}

// ---------------- kernel 1: fused conv2d(3x3,1->32)+ReLU + projection GEMM ----------------
// Workgroup tile: 32 rows (s) x 256 cols (h). Each wave: 32x32 register tile
// (2 A-frags x 2 B-frags -> 4 WMMAs), halving per-WMMA L2/B and LDS/A traffic
// vs a 16-row tile.
#define LDA 264   // bf16 row stride (256 + 8), keeps 16B alignment, staggers banks

__global__ __launch_bounds__(256) void k_feat_gemm(const float* __restrict__ x,
                                                   const float* __restrict__ c2w,
                                                   const float* __restrict__ c2b,
                                                   const float* __restrict__ projb,
                                                   const bf16_t* __restrict__ Bp,
                                                   float* __restrict__ feat) {
    __shared__ __align__(16) float  xp[34][260];       // rows s0-1..s0+32, cols -1..256
    __shared__ __align__(16) bf16_t A[32 * LDA];       // one conv channel: 32 x 256 bf16

    const int b  = blockIdx.y;
    const int s0 = blockIdx.x * 32;
    const int tid = threadIdx.x;

    for (int idx = tid; idx < 34 * 258; idx += 256) {
        int r = idx / 258, cc = idx % 258;
        int srow = s0 - 1 + r, dcol = cc - 1;
        float v = 0.f;
        if (srow >= 0 && srow < SS && dcol >= 0 && dcol < DD)
            v = x[((size_t)b * SS + srow) * DD + dcol];
        xp[r][cc] = v;
    }
    __syncthreads();

    const int lane = tid & 31, wid = tid >> 5;
    const int m = lane & 15, half = lane >> 4;
    const int nA = wid * 32 + (lane & 15);   // wave's first n-tile column
    const int nB = nA + 16;                  // wave's second n-tile column

    v8f acc00 = {}; v8f acc01 = {};          // rows 0-15   x {nA-tile, nB-tile}
    v8f acc10 = {}; v8f acc11 = {};          // rows 16-31  x {nA-tile, nB-tile}

    for (int c = 0; c < 32; ++c) {
        // --- conv channel c: thread = column d, 32 rows each ---
        {
            const float* wc = c2w + c * 9;
            float w00 = wc[0], w01 = wc[1], w02 = wc[2];
            float w10 = wc[3], w11 = wc[4], w12 = wc[5];
            float w20 = wc[6], w21 = wc[7], w22 = wc[8];
            float bias = c2b[c];
            int d = tid;
            #pragma unroll
            for (int mm = 0; mm < 32; ++mm) {
                float s = bias
                    + w00 * xp[mm + 0][d + 0] + w01 * xp[mm + 0][d + 1] + w02 * xp[mm + 0][d + 2]
                    + w10 * xp[mm + 1][d + 0] + w11 * xp[mm + 1][d + 1] + w12 * xp[mm + 1][d + 2]
                    + w20 * xp[mm + 2][d + 0] + w21 * xp[mm + 2][d + 1] + w22 * xp[mm + 2][d + 2];
                s = s > 0.f ? s : 0.f;
                A[mm * LDA + d] = (bf16_t)s;
            }
        }
        __syncthreads();

        // --- WMMA over this channel's K = 256 ---
        #pragma unroll
        for (int k0 = 0; k0 < 8; ++k0) {
            int kb8 = k0 * 32 + half * 8;
            // A frags: documented 16-bit A layout — two runs of 8 per lane
            v16bf a0 = ld_frag2(&A[m * LDA + kb8], &A[m * LDA + kb8 + 16]);
            v16bf a1 = ld_frag2(&A[(m + 16) * LDA + kb8], &A[(m + 16) * LDA + kb8 + 16]);
            int kb = c * 8 + k0;
            // B frags: two runs of 8 per lane (half*8.., 16+half*8..), matching A
            const bf16_t* bpA = Bp + (((size_t)kb * HH + nA) * 32 + half * 8);
            const bf16_t* bpB = Bp + (((size_t)kb * HH + nB) * 32 + half * 8);
            v16bf bA = ld_frag2(bpA, bpA + 16);
            v16bf bB = ld_frag2(bpB, bpB + 16);
            acc00 = wmma_bf16(a0, bA, acc00);
            acc01 = wmma_bf16(a0, bB, acc01);
            acc10 = wmma_bf16(a1, bA, acc10);
            acc11 = wmma_bf16(a1, bB, acc11);
        }
        __syncthreads();
    }

    // epilogue: + proj_b, store feat tile (32 x 256)
    float pbA = projb[nA], pbB = projb[nB];
    #pragma unroll
    for (int r = 0; r < 8; ++r) {
        int mm = (half ? 8 : 0) + r;
        size_t row0 = ((size_t)b * SS + s0 + mm) * HH;
        size_t row1 = ((size_t)b * SS + s0 + 16 + mm) * HH;
        feat[row0 + nA] = acc00[r] + pbA;
        feat[row0 + nB] = acc01[r] + pbB;
        feat[row1 + nA] = acc10[r] + pbA;
        feat[row1 + nB] = acc11[r] + pbB;
    }
}

// ---------------- kernel 2: conv1d(k3)+ReLU+fc+sigmoid as K=768 WMMA GEMM ----------------
#define LDF 264

__global__ __launch_bounds__(256) void k_prob(const float* __restrict__ feat,
                                              const float* __restrict__ c1b,
                                              const float* __restrict__ fcw,
                                              const float* __restrict__ fcb,
                                              const bf16_t* __restrict__ Bp2,
                                              float* __restrict__ prob) {
    __shared__ __align__(16) bf16_t fl[18 * LDF];     // feat rows s0-1..s0+16 in bf16
    __shared__ float rowsum[16];

    const int b  = blockIdx.y;
    const int s0 = blockIdx.x * 16;
    const int tid = threadIdx.x;

    if (tid < 16) rowsum[tid] = 0.f;
    for (int idx = tid; idx < 18 * 256; idx += 256) {
        int r = idx >> 8, hh = idx & 255;
        int srow = s0 - 1 + r;
        float v = (srow >= 0 && srow < SS) ? feat[((size_t)b * SS + srow) * HH + hh] : 0.f;
        fl[r * LDF + hh] = (bf16_t)v;
    }
    __syncthreads();

    const int lane = tid & 31, wid = tid >> 5;
    const int m = lane & 15, half = lane >> 4;
    const int nA = wid * 32 + (lane & 15);
    const int nB = nA + 16;

    v8f acc0 = {}; v8f acc1 = {};

    #pragma unroll 4
    for (int k0 = 0; k0 < 24; ++k0) {
        int kb8a = k0 * 32 + half * 8;
        int kb8b = kb8a + 16;
        int j = kb8a >> 8;                       // same j for both runs (32 | 256)
        v16bf afrag = ld_frag2(&fl[(m + j) * LDF + (kb8a & 255)],
                               &fl[(m + j) * LDF + (kb8b & 255)]);
        {
            const bf16_t* bp = Bp2 + (((size_t)k0 * HH + nA) * 32 + half * 8);
            acc0 = wmma_bf16(afrag, ld_frag2(bp, bp + 16), acc0);
        }
        {
            const bf16_t* bp = Bp2 + (((size_t)k0 * HH + nB) * 32 + half * 8);
            acc1 = wmma_bf16(afrag, ld_frag2(bp, bp + 16), acc1);
        }
    }

    // epilogue: + conv1d_b, ReLU, * fc_w[n], reduce over n (256 cols)
    float bA = c1b[nA], bB = c1b[nB], fA = fcw[nA], fB = fcw[nB];
    #pragma unroll
    for (int r = 0; r < 8; ++r) {
        float cA = acc0[r] + bA; cA = cA > 0.f ? cA : 0.f;
        float cB = acc1[r] + bB; cB = cB > 0.f ? cB : 0.f;
        float p = cA * fA + cB * fB;
        p += __shfl_xor(p, 1, 32);
        p += __shfl_xor(p, 2, 32);
        p += __shfl_xor(p, 4, 32);
        p += __shfl_xor(p, 8, 32);
        if ((lane & 15) == 0) atomicAdd(&rowsum[(half ? 8 : 0) + r], p);
    }
    __syncthreads();
    if (tid < 16) {
        float sv = rowsum[tid] + fcb[0];
        prob[(size_t)b * SS + s0 + tid] = 1.f / (1.f + expf(-sv));
    }
}

// ---------------- kernel 3: CIF scan + deterministic monotone scatter ----------------
__global__ __launch_bounds__(256) void k_cif(const float* __restrict__ prob,
                                             const int* __restrict__ xlen,
                                             const float* __restrict__ feat,
                                             float* __restrict__ out,
                                             float* __restrict__ absdiff) {
    __shared__ float al[SS];     // raw masked alpha
    __shared__ float cs[SS];     // inclusive cumsum of raw alpha
    __shared__ float wsum[9];

    const int b = blockIdx.x, tid = threadIdx.x;
    const int L = xlen[b];
    const int tgt = L / 4;                         // DS = 4

    // load alpha (mask pads), per-thread 8 elements
    float loc[8]; float s = 0.f;
    const int base = tid * 8;
    #pragma unroll
    for (int i = 0; i < 8; ++i) {
        int ss = base + i;
        float v = (ss < L) ? prob[(size_t)b * SS + ss] : 0.f;
        loc[i] = v; s += v;
    }
    // block-wide exclusive scan of per-thread sums
    const int lane = tid & 31, wid = tid >> 5;
    float v = s;
    for (int off = 1; off < 32; off <<= 1) {
        float nv = __shfl_up(v, off, 32);
        if (lane >= off) v += nv;
    }
    if (lane == 31) wsum[wid] = v;
    __syncthreads();
    if (tid == 0) {
        float r = 0.f;
        for (int w = 0; w < 8; ++w) { float t = wsum[w]; wsum[w] = r; r += t; }
        wsum[8] = r;
    }
    __syncthreads();
    float run = wsum[wid] + (v - s);               // exclusive prefix for this thread
    #pragma unroll
    for (int i = 0; i < 8; ++i) { run += loc[i]; cs[base + i] = run; al[base + i] = loc[i]; }
    __syncthreads();

    const float alpha_sum = wsum[8];
    const float scale = ((float)tgt + 1e-4f) / alpha_sum;

    // scatter: thread owns channel c; target stream is monotone non-decreasing,
    // so one register accumulator + flush-on-change is race-free and deterministic.
    const int c = tid;
    const float* fb = feat + (size_t)b * SS * HH;
    float* ob = out + (size_t)b * TT * HH;
    float acc = 0.f; int tcur = 0;

    for (int ss = 0; ss < SS; ++ss) {
        float c1v = cs[ss] * scale;
        float c0v = ss ? cs[ss - 1] * scale : 0.f;
        int ridx = (int)floorf(c1v); ridx = ridx > TT ? TT : ridx;
        int lidx = (int)floorf(c0v); lidx = lidx > TT ? TT : lidx;
        int fire = ridx - lidx;
        int extra = fire - 1; if (extra < 0) extra = 0;
        float a  = al[ss] * scale;
        float rw = (fire > 0) ? (c1v - (float)ridx) : 0.f;
        float lw = a - rw - (float)extra;
        float f  = fb[(size_t)ss * HH + c];

        {   // left weight -> lidx
            float vv = lw * f;
            if (lidx == tcur) acc += vv;
            else { if (tcur < TT) ob[(size_t)tcur * HH + c] = acc; tcur = lidx; acc = vv; }
        }
        #pragma unroll
        for (int e = 1; e <= 4; ++e) {             // MAX_EXTRA, src_feats = feat * BETA(=1)
            if (extra >= e) {
                int t = lidx + e; if (t > TT) t = TT;
                if (t == tcur) acc += f;
                else { if (tcur < TT) ob[(size_t)tcur * HH + c] = acc; tcur = t; acc = f; }
            }
        }
        if (fire > 0) {                            // right weight -> ridx
            float vv = rw * f;
            if (ridx == tcur) acc += vv;
            else { if (tcur < TT) ob[(size_t)tcur * HH + c] = acc; tcur = ridx; acc = vv; }
        }
    }
    if (tcur < TT) ob[(size_t)tcur * HH + c] = acc;

    if (tid == 0) absdiff[b] = fabsf(alpha_sum - (float)tgt);
}

// ---------------- kernel 4: tgt_len + loss ----------------
__global__ void k_final(const int* __restrict__ xlen, const float* __restrict__ absdiff,
                        float* __restrict__ outTg, float* __restrict__ outL) {
    int t = threadIdx.x;
    if (t < BB) outTg[t] = (float)(xlen[t] / 4);
    if (t == 0) outL[0] = 0.25f * (absdiff[0] + absdiff[1] + absdiff[2] + absdiff[3]);
}

// ---------------- launch ----------------
extern "C" void kernel_launch(void* const* d_in, const int* in_sizes, int n_in,
                              void* d_out, int out_size, void* d_ws, size_t ws_size,
                              hipStream_t stream) {
    (void)in_sizes; (void)n_in; (void)out_size; (void)ws_size;

    const float* x     = (const float*)d_in[0];
    const int*   x_len = (const int*)  d_in[1];
    const float* c2w   = (const float*)d_in[2];
    const float* c2b   = (const float*)d_in[3];
    const float* projw = (const float*)d_in[4];
    const float* projb = (const float*)d_in[5];
    const float* c1w   = (const float*)d_in[6];
    const float* c1b   = (const float*)d_in[7];
    const float* fcw   = (const float*)d_in[8];
    const float* fcb   = (const float*)d_in[9];

    float* out = (float*)d_out;
    float* outO  = out;                    // (4,512,256) = 524288
    float* outTg = out + 524288;           // (4,)
    float* outL  = out + 524292;           // scalar
    float* outP  = out + 524293;           // (4,2048) = 8192

    char* ws = (char*)d_ws;
    bf16_t* B1   = (bf16_t*)ws;                                   // 4 MB
    bf16_t* B2   = (bf16_t*)(ws + 4194304);                       // 384 KB
    float*  feat = (float*)(ws + 4194304 + 393216);               // 8 MB
    float*  absd = (float*)(ws + 4194304 + 393216 + 8388608);     // 16 B

    k_pack<<<(K1 * HH + K2 * HH) / 256, 256, 0, stream>>>(projw, c1w, B1, B2);
    k_zero<<<524288 / 256, 256, 0, stream>>>(outO, 524288);

    dim3 grid1(SS / 32, BB);
    k_feat_gemm<<<grid1, 256, 0, stream>>>(x, c2w, c2b, projb, B1, feat);
    dim3 grid2(SS / 16, BB);
    k_prob<<<grid2, 256, 0, stream>>>(feat, c1b, fcw, fcb, B2, outP);
    k_cif<<<BB, 256, 0, stream>>>(outP, x_len, feat, outO, absd);
    k_final<<<1, 64, 0, stream>>>(x_len, absd, outTg, outL);
}
